// MultiHeadTrendAwareCrossAttention_1391569404155
// MI455X (gfx1250) — matrix operand order repaired and checked by reading, not compiled, not run on hardware
//
#include <hip/hip_runtime.h>

typedef __bf16 bf16;
typedef __attribute__((ext_vector_type(16))) __bf16 v16bf;
typedef __attribute__((ext_vector_type(8)))  __bf16 v8bf;
typedef __attribute__((ext_vector_type(8)))  float  v8f;
typedef __attribute__((ext_vector_type(4)))  unsigned int u32x4;
typedef __attribute__((ext_vector_type(8)))  int i32x8;
typedef __attribute__((ext_vector_type(4)))  int i32x4;

#define B_  4
#define L_  2048
#define D_  512
#define H_  8
#define HD_ 64

// ---------------- WMMA helpers ----------------

static __device__ inline v8f wmma_bf(v16bf a, v16bf b, v8f c) {
  return __builtin_amdgcn_wmma_f32_16x16x32_bf16(false, a, false, b, (short)0, c,
                                                 false, false);
}

// A-fragment (16x32 bf16, row-major source): lane = M, half 0 -> K{0..7,16..23},
// half 1 -> K{8..15,24..31}.  p points at row start + kb.
static __device__ inline v16bf frag_a(const bf16* p, int half) {
  v8bf x = *(const v8bf*)(p + half * 8);
  v8bf y = *(const v8bf*)(p + half * 8 + 16);
  v16bf r;
#pragma unroll
  for (int i = 0; i < 8; ++i) { r[i] = x[i]; r[i + 8] = y[i]; }
  return r;
}

// B-fragment (32x16 bf16): lane = N holds one column; half 0 -> K{0..7,16..23},
// half 1 -> K{8..15,24..31}.  p points at Bt row (column of B) start + kb.
static __device__ inline v16bf frag_b(const bf16* p, int half) {
  v8bf x = *(const v8bf*)(p + half * 8);
  v8bf y = *(const v8bf*)(p + half * 8 + 16);
  v16bf r;
#pragma unroll
  for (int i = 0; i < 8; ++i) { r[i] = x[i]; r[i + 8] = y[i]; }
  return r;
}

// A-fragment for the conv1d-as-GEMM: virtual A row = [x[l-1] | x[l] | x[l+1]],
// K-dim = 3*512, zeros outside the sequence.
static __device__ inline v16bf frag_a_conv(const bf16* Abatch, int l, int kb, int half) {
  v16bf r;
#pragma unroll
  for (int seg = 0; seg < 2; ++seg) {
    int kk = kb + half * 8 + seg * 16;
    int k  = kk >> 9;          // which tap (0..2)
    int i  = kk & 511;         // channel
    int sl = l + k - 1;
    v8bf v;
    if (sl >= 0 && sl < L_) {
      v = *(const v8bf*)(Abatch + (size_t)sl * D_ + i);
    } else {
#pragma unroll
      for (int j = 0; j < 8; ++j) v[j] = (bf16)0.0f;
    }
#pragma unroll
    for (int j = 0; j < 8; ++j) r[seg * 8 + j] = v[j];
  }
  return r;
}

// ---------------- DPP 16-lane butterfly reductions (no LDS traffic) --------

template <int CTRL>
static __device__ inline float dpp_f(float v) {
  return __int_as_float(__builtin_amdgcn_update_dpp(
      0, __float_as_int(v), CTRL, 0xF, 0xF, true));
}

static __device__ inline float red_max16(float v) {
  v = fmaxf(v, dpp_f<0xB1>(v));    // quad_perm(1,0,3,2)
  v = fmaxf(v, dpp_f<0x4E>(v));    // quad_perm(2,3,0,1)
  v = fmaxf(v, dpp_f<0x141>(v));   // row_half_mirror
  v = fmaxf(v, dpp_f<0x140>(v));   // row_mirror
  return v;
}

static __device__ inline float red_sum16(float v) {
  v += dpp_f<0xB1>(v);
  v += dpp_f<0x4E>(v);
  v += dpp_f<0x141>(v);
  v += dpp_f<0x140>(v);
  return v;
}

// ---------------- Tensor Data Mover: 2-D tile -> LDS ----------------
// D# per CDNA5 ISA.  data_size=2B, count=1, type=2,
// tile (tile_w x tile_h) elements, global row stride = row_stride_elems.
static __device__ inline void tdm_load_2d(unsigned lds_addr, const void* gptr,
                                          unsigned tile_w, unsigned tile_h,
                                          unsigned row_stride_elems) {
  unsigned long long ga = (unsigned long long)(size_t)gptr;
  u32x4 g0;
  g0[0] = 1u;                                              // count=1, user D#
  g0[1] = lds_addr;                                        // LDS byte address
  g0[2] = (unsigned)ga;                                    // global addr [31:0]
  g0[3] = (unsigned)((ga >> 32) & 0x01ffffffu) | (2u << 30); // addr[56:32] | type=2
  i32x8 g1;
  g1[0] = (int)(1u << 16);                                 // data_size=1 -> 2 bytes
  g1[1] = (int)(tile_w << 16);                             // tensor_dim0[15:0]
  g1[2] = (int)(tile_h << 16);                             // tensor_dim1[15:0]
  g1[3] = (int)(tile_w << 16);                             // tile_dim0
  g1[4] = (int)tile_h;                                     // tile_dim1 (tile_dim2=0)
  g1[5] = (int)row_stride_elems;                           // tensor_dim0_stride[31:0]
  g1[6] = 0;
  g1[7] = 0;
  i32x4 z4 = {0, 0, 0, 0};
  i32x8 z8 = {0, 0, 0, 0, 0, 0, 0, 0};
  __builtin_amdgcn_tensor_load_to_lds(g0, g1, z4, z4, z8, 0);
}

// ---------------- prep kernels ----------------

__global__ __launch_bounds__(128) void ln_kernel(const float* __restrict__ x,
                                                 const float* __restrict__ g,
                                                 const float* __restrict__ bb,
                                                 bf16* __restrict__ y) {
  int row = blockIdx.x;                     // B*L rows
  const float* xr = x + (size_t)row * D_;
  int tid = threadIdx.x;
  float v[4], s = 0.f, s2 = 0.f;
#pragma unroll
  for (int i = 0; i < 4; ++i) {
    v[i] = xr[tid + i * 128];
    s += v[i]; s2 += v[i] * v[i];
  }
  __shared__ float rs[128], rq[128];
  rs[tid] = s; rq[tid] = s2;
  __syncthreads();
  for (int off = 64; off; off >>= 1) {
    if (tid < off) { rs[tid] += rs[tid + off]; rq[tid] += rq[tid + off]; }
    __syncthreads();
  }
  float mu  = rs[0] * (1.0f / D_);
  float var = rq[0] * (1.0f / D_) - mu * mu;
  float inv = rsqrtf(var + 1e-5f);
#pragma unroll
  for (int i = 0; i < 4; ++i) {
    int c = tid + i * 128;
    y[(size_t)row * D_ + c] = (bf16)((v[i] - mu) * inv * g[c] + bb[c]);
  }
}

__global__ __launch_bounds__(256) void cvt_kernel(const float* __restrict__ x,
                                                  bf16* __restrict__ y, int n) {
  int i = blockIdx.x * 256 + threadIdx.x;
  if (i < n) y[i] = (bf16)x[i];
}

// dst[n*M + m] = src[m*N + n]  (convert f32 -> bf16)
__global__ __launch_bounds__(256) void transpose_kernel(const float* __restrict__ src,
                                                        bf16* __restrict__ dst,
                                                        int M, int N) {
  int i = blockIdx.x * 256 + threadIdx.x;
  if (i < M * N) {
    int m = i / N, n = i - m * N;
    dst[(size_t)n * M + m] = (bf16)src[i];
  }
}

// ---------------- GEMM kernel ----------------
// Block = (64*MT) rows x 64 cols; each of 4 waves owns MT M-tiles so every B
// fragment load feeds MT WMMAs.  EPI: 0 = bf16 row-major, 1 = f32 row-major
// (final output), 2 = scatter into per-head transposed V layout.
template <bool CONV, int EPI, int MT>
__global__ __launch_bounds__(128) void gemm_kernel(const bf16* __restrict__ A,
                                                   const bf16* __restrict__ Bt,
                                                   const float* __restrict__ bias,
                                                   void* __restrict__ Cv) {
  const int Kdim = CONV ? 3 * D_ : D_;
  const int lane = threadIdx.x & 31;
  const int wave = threadIdx.x >> 5;
  const int ln15 = lane & 15;
  const int half = lane >> 4;
  const int rb   = blockIdx.x * (64 * MT) + wave * (16 * MT);   // wave row base
  const int n0   = blockIdx.y * 64;
  const int bi   = (blockIdx.x * (64 * MT)) >> 11;              // batch (uniform)
  const bf16* Abase = A + (size_t)bi * L_ * D_;

  int lm[MT];
  const bf16* Arm[MT];
#pragma unroll
  for (int mt = 0; mt < MT; ++mt) {
    lm[mt]  = (rb + mt * 16 + ln15) & (L_ - 1);
    Arm[mt] = A + (size_t)(rb + mt * 16 + ln15) * D_;
  }

  v8f acc[MT][4];
#pragma unroll
  for (int mt = 0; mt < MT; ++mt)
#pragma unroll
    for (int nt = 0; nt < 4; ++nt)
#pragma unroll
      for (int r = 0; r < 8; ++r) acc[mt][nt][r] = 0.0f;

  for (int kb = 0; kb < Kdim; kb += 32) {
    v16bf a[MT];
#pragma unroll
    for (int mt = 0; mt < MT; ++mt)
      a[mt] = CONV ? frag_a_conv(Abase, lm[mt], kb, half) : frag_a(Arm[mt] + kb, half);
#pragma unroll
    for (int nt = 0; nt < 4; ++nt) {
      v16bf b = frag_b(Bt + (size_t)(n0 + nt * 16 + ln15) * Kdim + kb, half);
#pragma unroll
      for (int mt = 0; mt < MT; ++mt)
        acc[mt][nt] = wmma_bf(a[mt], b, acc[mt][nt]);
    }
  }

#pragma unroll
  for (int mt = 0; mt < MT; ++mt)
#pragma unroll
    for (int nt = 0; nt < 4; ++nt) {
      int n = n0 + nt * 16 + ln15;
      float bv = bias[n];
#pragma unroll
      for (int r = 0; r < 8; ++r) {
        int row = rb + mt * 16 + r + 8 * half;
        float val = acc[mt][nt][r] + bv;
        if (EPI == 0) {
          ((bf16*)Cv)[(size_t)row * D_ + n] = (bf16)val;
        } else if (EPI == 1) {
          ((float*)Cv)[(size_t)row * D_ + n] = val;
        } else {
          int l2 = row & (L_ - 1);
          int hh = n >> 6, d = n & 63;
          ((bf16*)Cv)[((size_t)(bi * H_ + hh) * HD_ + d) * L_ + l2] = (bf16)val;
        }
      }
    }
}

// ---------------- flash attention ----------------
// One block = (b, h, 128-query tile); 8 waves x 16 queries share each K/V tile
// (halves K/V L2 traffic and amortizes TDM issue + barrier over 2x compute).
// K/V 64-key chunks stream into ping-pong LDS buffers via the Tensor Data
// Mover; the next chunk's DMA runs underneath the current chunk's compute.
// Online softmax in log2-domain with DPP reductions.
__global__ __launch_bounds__(256) void attn_kernel(const bf16* __restrict__ Q,
                                                   const bf16* __restrict__ Km,
                                                   const bf16* __restrict__ Vt,
                                                   bf16* __restrict__ O) {
  const int idx  = blockIdx.x;
  const int qb   = idx & 15;          // L/128
  const int h    = (idx >> 4) & 7;
  const int b    = idx >> 7;
  const int lane = threadIdx.x & 31;
  const int wave = threadIdx.x >> 5;  // 0..7
  const int ln15 = lane & 15;
  const int half = lane >> 4;

  __shared__ bf16 lds_k[2][64][64];   // ping-pong [key][d]
  __shared__ bf16 lds_v[2][64][64];   // ping-pong [d][key]
  __shared__ bf16 lds_p[8][16][64];   // per-wave P staging [q][key]

  const bf16* Qbase = Q + ((size_t)(b * L_) + qb * 128 + wave * 16) * D_ + h * HD_;
  const bf16* Kbase = Km + (size_t)(b * L_) * D_ + h * HD_;
  const bf16* Vbase = Vt + (size_t)(b * H_ + h) * HD_ * L_;
  unsigned ldsk[2], ldsv[2];
  ldsk[0] = (unsigned)(size_t)&lds_k[0][0][0];
  ldsk[1] = (unsigned)(size_t)&lds_k[1][0][0];
  ldsv[0] = (unsigned)(size_t)&lds_v[0][0][0];
  ldsv[1] = (unsigned)(size_t)&lds_v[1][0][0];

  // Q fragments, pre-scaled by HD^-0.5 * log2(e) so scores are already in the
  // exp2 domain and need no per-chunk multiply.
  const float sc = 0.125f * 1.44269504f;
  v16bf qa0 = frag_a(Qbase + (size_t)ln15 * D_ + 0, half);
  v16bf qa1 = frag_a(Qbase + (size_t)ln15 * D_ + 32, half);
#pragma unroll
  for (int i = 0; i < 16; ++i) {
    qa0[i] = (bf16)((float)qa0[i] * sc);
    qa1[i] = (bf16)((float)qa1[i] * sc);
  }

  float mrun[8], lrun[8];
  v8f acc[4];
#pragma unroll
  for (int r = 0; r < 8; ++r) { mrun[r] = -3.0e38f; lrun[r] = 0.0f; }
#pragma unroll
  for (int nt = 0; nt < 4; ++nt)
#pragma unroll
    for (int r = 0; r < 8; ++r) acc[nt][r] = 0.0f;

  // prologue: start chunk 0 DMA (K by wave 0, V by wave 1)
  if (wave == 0)      tdm_load_2d(ldsk[0], Kbase, HD_, 64, D_);
  else if (wave == 1) tdm_load_2d(ldsv[0], Vbase, 64, HD_, L_);

  const int NCH = L_ / 64;
  for (int ic = 0; ic < NCH; ++ic) {
    const int buf = ic & 1;
    __builtin_amdgcn_s_wait_tensorcnt(0);   // own DMA done (no-op for others)
    __syncthreads();                        // tile published; prev buf free

    // launch next chunk's DMA into the other buffer; overlaps compute below
    if (ic + 1 < NCH) {
      const int kcn = (ic + 1) * 64;
      if (wave == 0)      tdm_load_2d(ldsk[buf ^ 1], Kbase + (size_t)kcn * D_, HD_, 64, D_);
      else if (wave == 1) tdm_load_2d(ldsv[buf ^ 1], Vbase + kcn, 64, HD_, L_);
    }

    const bf16 (*kt)[64] = lds_k[buf];
    const bf16 (*vt)[64] = lds_v[buf];

    // scores: S(16x64) = Qs(16x64) @ K^T   (already in exp2 domain)
    v8f s[4];
#pragma unroll
    for (int t = 0; t < 4; ++t) {
      v8f c;
#pragma unroll
      for (int r = 0; r < 8; ++r) c[r] = 0.0f;
      v16bf kb0 = frag_b(&kt[t * 16 + ln15][0], half);
      v16bf kb1 = frag_b(&kt[t * 16 + ln15][32], half);
      c = wmma_bf(qa0, kb0, c);
      c = wmma_bf(qa1, kb1, c);
      s[t] = c;
    }

    // chunk row-max via DPP butterfly (stays inside each 16-lane half)
    float mc[8];
#pragma unroll
    for (int r = 0; r < 8; ++r) {
      mc[r] = fmaxf(fmaxf(s[0][r], s[1][r]), fmaxf(s[2][r], s[3][r]));
      mc[r] = red_max16(mc[r]);
    }

    float corr[8], lc[8];
#pragma unroll
    for (int r = 0; r < 8; ++r) {
      float mn = fmaxf(mrun[r], mc[r]);
      corr[r]  = __builtin_amdgcn_exp2f(mrun[r] - mn);
      mrun[r]  = mn;
      lc[r]    = 0.0f;
    }
#pragma unroll
    for (int t = 0; t < 4; ++t)
#pragma unroll
      for (int r = 0; r < 8; ++r) {
        float p = __builtin_amdgcn_exp2f(s[t][r] - mrun[r]);
        s[t][r] = p;
        lc[r] += p;
      }
#pragma unroll
    for (int r = 0; r < 8; ++r) {
      lc[r]   = red_sum16(lc[r]);
      lrun[r] = lrun[r] * corr[r] + lc[r];
    }
#pragma unroll
    for (int nt = 0; nt < 4; ++nt)
#pragma unroll
      for (int r = 0; r < 8; ++r) acc[nt][r] *= corr[r];

    // stage P through per-wave LDS to re-layout C-fragment -> A-fragment
#pragma unroll
    for (int t = 0; t < 4; ++t)
#pragma unroll
      for (int r = 0; r < 8; ++r)
        lds_p[wave][r + 8 * half][t * 16 + ln15] = (bf16)s[t][r];

    v16bf pa0 = frag_a(&lds_p[wave][ln15][0], half);
    v16bf pa1 = frag_a(&lds_p[wave][ln15][32], half);

    // O(16x64) += P(16x64) @ V(64x64)
#pragma unroll
    for (int nt = 0; nt < 4; ++nt) {
      v16bf vb0 = frag_b(&vt[nt * 16 + ln15][0], half);
      v16bf vb1 = frag_b(&vt[nt * 16 + ln15][32], half);
      acc[nt] = wmma_bf(pa0, vb0, acc[nt]);
      acc[nt] = wmma_bf(pa1, vb1, acc[nt]);
    }
  }

  bf16* Obase = O + ((size_t)(b * L_) + qb * 128 + wave * 16) * D_ + h * HD_;
#pragma unroll
  for (int r = 0; r < 8; ++r) {
    float inv = 1.0f / lrun[r];
#pragma unroll
    for (int nt = 0; nt < 4; ++nt) {
      float v = acc[nt][r] * inv;
      Obase[(size_t)(r + 8 * half) * D_ + nt * 16 + ln15] = (bf16)v;
    }
  }
}

// ---------------- host launcher ----------------

extern "C" void kernel_launch(void* const* d_in, const int* in_sizes, int n_in,
                              void* d_out, int out_size, void* d_ws, size_t ws_size,
                              hipStream_t stream) {
  const float* query = (const float*)d_in[0];
  const float* keyv  = (const float*)d_in[1];
  const float* ln_g  = (const float*)d_in[2];
  const float* ln_b  = (const float*)d_in[3];
  const float* Wq    = (const float*)d_in[4];
  const float* bq    = (const float*)d_in[5];
  const float* Wk    = (const float*)d_in[6];
  const float* bk    = (const float*)d_in[7];
  const float* Wv    = (const float*)d_in[8];
  const float* bv    = (const float*)d_in[9];
  const float* Wo    = (const float*)d_in[10];
  const float* bo    = (const float*)d_in[11];
  float* out = (float*)d_out;

  const size_t NE = (size_t)B_ * L_ * D_;   // 4,194,304 elements
  char* ws = (char*)d_ws;
  size_t off = 0;
  auto alloc = [&](size_t bytes) { size_t o = off; off += (bytes + 255) & ~(size_t)255; return o; };

  bf16* qn  = (bf16*)(ws + alloc(NE * 2));
  bf16* kvb = (bf16*)(ws + alloc(NE * 2));
  bf16* Qb  = (bf16*)(ws + alloc(NE * 2));
  bf16* Kb  = (bf16*)(ws + alloc(NE * 2));
  bf16* Vtb = (bf16*)(ws + alloc(NE * 2));
  bf16* Ob  = (bf16*)(ws + alloc(NE * 2));
  bf16* Wqt = (bf16*)(ws + alloc((size_t)3 * D_ * D_ * 2));
  bf16* Wkt = (bf16*)(ws + alloc((size_t)3 * D_ * D_ * 2));
  bf16* Wvt = (bf16*)(ws + alloc((size_t)D_ * D_ * 2));
  bf16* Wot = (bf16*)(ws + alloc((size_t)D_ * D_ * 2));
  if (off > ws_size) return;

  // prep
  ln_kernel<<<B_ * L_, 128, 0, stream>>>(query, ln_g, ln_b, qn);
  cvt_kernel<<<(int)(NE / 256), 256, 0, stream>>>(keyv, kvb, (int)NE);
  {
    int mw = 3 * D_, n = D_;
    transpose_kernel<<<(mw * n + 255) / 256, 256, 0, stream>>>(Wq, Wqt, mw, n);
    transpose_kernel<<<(mw * n + 255) / 256, 256, 0, stream>>>(Wk, Wkt, mw, n);
    transpose_kernel<<<(D_ * D_ + 255) / 256, 256, 0, stream>>>(Wv, Wvt, D_, D_);
    transpose_kernel<<<(D_ * D_ + 255) / 256, 256, 0, stream>>>(Wo, Wot, D_, D_);
  }

  dim3 ggc(B_ * L_ / 128, D_ / 64);   // conv GEMMs: MT=2
  dim3 ggd(B_ * L_ / 256, D_ / 64);   // dense GEMMs: MT=4
  // Q = conv1d(qn, Wq) + bq          (bf16 row-major)
  gemm_kernel<true, 0, 2><<<ggc, 128, 0, stream>>>(qn, Wqt, bq, Qb);
  // K = conv1d(kv, Wk) + bk          (bf16 row-major)
  gemm_kernel<true, 0, 2><<<ggc, 128, 0, stream>>>(kvb, Wkt, bk, Kb);
  // V = kv @ Wv + bv                 (scattered per-head transposed)
  gemm_kernel<false, 2, 4><<<ggd, 128, 0, stream>>>(kvb, Wvt, bv, Vtb);

  // attention: 512 blocks x 8 waves (128 queries per block)
  attn_kernel<<<B_ * H_ * (L_ / 128), 256, 0, stream>>>(Qb, Kb, Vtb, Ob);

  // out = O @ Wo + bo                (f32)
  gemm_kernel<false, 1, 4><<<ggd, 128, 0, stream>>>(Ob, Wot, bo, out);

  (void)in_sizes; (void)n_in; (void)out_size;
}
